// NoncommutativeKolmogorovArnoldOperator_16441134809573
// MI455X (gfx1250) — compile-verified, matrix operand order checked by reading, not run on hardware
//
#include <hip/hip_runtime.h>
#include <hip/hip_bf16.h>
#include <math.h>
#include <stdint.h>

#define DIM 2048

typedef __attribute__((ext_vector_type(16))) _Float16 v16h;
typedef __attribute__((ext_vector_type(8)))  _Float16 v8h;
typedef __attribute__((ext_vector_type(8)))  float    v8f;

// TDM descriptor vector types
typedef unsigned int tdm_v4u __attribute__((ext_vector_type(4)));
typedef int          tdm_v8i __attribute__((ext_vector_type(8)));
typedef int          tdm_v4i __attribute__((ext_vector_type(4)));

#if defined(__has_builtin)
#if __has_builtin(__builtin_amdgcn_tensor_load_to_lds)
#define HAVE_TDM 1
#endif
#endif
#ifndef HAVE_TDM
#define HAVE_TDM 0
#endif

// ---------------- d_out layout (in doubles) ----------------
#define OUT_M      0
#define OUT_EIG    (2*DIM*DIM)
#define OUT_CONN   (OUT_EIG + DIM)
#define OUT_TRACE  (OUT_CONN + 2*DIM*DIM)
#define OUT_DET    (OUT_TRACE + 2)
#define OUT_MET    (OUT_DET + 2)
#define OUT_TOTAL  (OUT_MET + 5)

// ---------------- workspace layout (bytes) ----------------
#define WS_MR16    ((size_t)0)                        // DIM*DIM f16 (8 MB)
#define WS_MI16    ((size_t)8388608)                  // DIM*DIM f16 (8 MB)
#define WS_VFRAG   ((size_t)16777216)                 // 65536 f16: Vr-frag | Vi-frag (128 KB)
#define WS_YR      (WS_VFRAG + (size_t)131072)        // DIM*16 f32
#define WS_YI      (WS_YR + (size_t)DIM*16*4)
#define WS_ROWSUM  (WS_YI + (size_t)DIM*16*4)         // DIM doubles
#define WS_SCAL    (WS_ROWSUM + (size_t)DIM*8)        // [0]=fro [1]=min|diag| [2]=rho

// ================= zero / init =================
__global__ void k_zero_out(double* out) {
  size_t i = (size_t)blockIdx.x * blockDim.x + threadIdx.x;
  size_t st = (size_t)gridDim.x * blockDim.x;
  for (; i < (size_t)OUT_TOTAL; i += st) out[i] = 0.0;
}

__global__ void k_ws_init(_Float16* Mr16, _Float16* Mi16, _Float16* Vf) {
  size_t i = (size_t)blockIdx.x * blockDim.x + threadIdx.x;
  size_t st = (size_t)gridDim.x * blockDim.x;
  for (size_t k = i; k < (size_t)DIM*DIM; k += st) {
    Mr16[k] = (_Float16)0.0f; Mi16[k] = (_Float16)0.0f;
  }
  for (size_t k = i; k < (size_t)32768; k += st) {
    Vf[k]         = (_Float16)0.0220971f;   // 1/sqrt(2048)
    Vf[32768 + k] = (_Float16)0.0f;
  }
}

// ================= exact banded build of M (fp64) =================
__global__ void k_build_band(const double* __restrict__ arn, const double* __restrict__ alg,
                             const double* sr_p, const double* si_p, const double* th_p,
                             double* __restrict__ out, _Float16* Mr16, _Float16* Mi16) {
  int t = blockIdx.x * blockDim.x + threadIdx.x;
  if (t >= DIM * 15) return;
  int i = t / 15;
  int d = t % 15 - 7;
  int j = i + d;
  if (j < 0 || j >= DIM) return;
  double sr = *sr_p, si = *si_p, th = *th_p;
  double sabs = sqrt(sr * sr + si * si);
  double mre = 0.0, mim = 0.0;
  for (int p = i - 1; p <= i + 1; ++p) {
    if (p < 0 || p >= DIM) continue;
    double aip = arn[2 * ((size_t)i * DIM + p)];
    for (int q = j - 1; q <= j + 1; ++q) {
      if (q < 0 || q >= DIM) continue;
      double ajq = arn[2 * ((size_t)j * DIM + q)];
      double rre, rim;
      if (p == q) {
        double ln = log((double)(p + 1));
        double mag = exp(-sr * ln);
        rre = mag * cos(si * ln);
        rim = -mag * sin(si * ln);
      } else {
        int dd = p > q ? p - q : q - p;
        if (dd > 5) continue;
        rre = th * exp(-(double)dd / 10.0);
        rim = 0.0;
      }
      double w = aip * ajq;
      mre += w * rre; mim += w * rim;
    }
  }
  mre += sabs * alg[2 * ((size_t)i * DIM + j)];
  mim += sabs * alg[2 * ((size_t)i * DIM + j) + 1];
  size_t idx = (size_t)i * DIM + j;
  out[OUT_M + 2 * idx]     = mre;
  out[OUT_M + 2 * idx + 1] = mim;
  Mr16[idx] = (_Float16)(float)mre;
  Mi16[idx] = (_Float16)(float)mim;
}

// ================= riemann connection (banded: |i-j| <= 8) =================
__global__ void k_conn(const double* __restrict__ out_in, double* __restrict__ out) {
  int t = blockIdx.x * blockDim.x + threadIdx.x;
  if (t >= DIM * 17) return;
  int i = t / 17;
  int d = t % 17 - 8;
  int j = i + d;
  if (i >= DIM - 1 || j < 0 || j >= DIM - 1) return;
  size_t a = (size_t)(i + 1) * DIM + j;
  size_t b = (size_t)i * DIM + (j + 1);
  size_t c = (size_t)i * DIM + j;
  out[OUT_CONN + 2 * c]     = out_in[OUT_M + 2 * a]     - out_in[OUT_M + 2 * b];
  out[OUT_CONN + 2 * c + 1] = out_in[OUT_M + 2 * a + 1] - out_in[OUT_M + 2 * b + 1];
}

// ================= per-row |M|^2 band sums =================
__global__ void k_row_stats(const double* __restrict__ out, double* __restrict__ rowsum) {
  int i = blockIdx.x * blockDim.x + threadIdx.x;
  if (i >= DIM) return;
  int j0 = i - 7 < 0 ? 0 : i - 7;
  int j1 = i + 7 >= DIM ? DIM - 1 : i + 7;
  double s = 0.0;
  for (int j = j0; j <= j1; ++j) {
    double re = out[OUT_M + 2 * ((size_t)i * DIM + j)];
    double im = out[OUT_M + 2 * ((size_t)i * DIM + j) + 1];
    s += re * re + im * im;
  }
  rowsum[i] = s;
}

// ================= trace, log-det, min|diag|, fro =================
__global__ void k_finalize_diag(const double* __restrict__ outM, const double* __restrict__ rowsum,
                                double* __restrict__ out, double* __restrict__ scal) {
  __shared__ double s0[256], s1[256], s2[256], s3[256], s4[256], s5[256];
  int t = threadIdx.x;
  double trre = 0, trim = 0, lg = 0, ar = 0, fro = 0, mn = 1e300;
  for (int i = t; i < DIM; i += 256) {
    double re = outM[OUT_M + 2 * ((size_t)i * DIM + i)];
    double im = outM[OUT_M + 2 * ((size_t)i * DIM + i) + 1];
    trre += re; trim += im;
    double a2 = re * re + im * im;
    lg += 0.5 * log(a2);
    ar += atan2(im, re);
    mn = fmin(mn, sqrt(a2));
    fro += rowsum[i];
  }
  s0[t] = trre; s1[t] = trim; s2[t] = lg; s3[t] = ar; s4[t] = fro; s5[t] = mn;
  __syncthreads();
  for (int off = 128; off > 0; off >>= 1) {
    if (t < off) {
      s0[t] += s0[t+off]; s1[t] += s1[t+off]; s2[t] += s2[t+off];
      s3[t] += s3[t+off]; s4[t] += s4[t+off]; s5[t] = fmin(s5[t], s5[t+off]);
    }
    __syncthreads();
  }
  if (t == 0) {
    out[OUT_TRACE]     = s0[0];
    out[OUT_TRACE + 1] = s1[0];
    double mag = exp(s2[0]);
    out[OUT_DET]     = mag * cos(s3[0]);
    out[OUT_DET + 1] = mag * sin(s3[0]);
    scal[0] = sqrt(s4[0]);
    scal[1] = s5[0];
  }
}

// ================= eigenvalues approx: bitonic sort of Re(diag) =================
__global__ void k_sort_eig(const double* __restrict__ outM, double* __restrict__ out) {
  __shared__ double s[DIM];
  for (int i = threadIdx.x; i < DIM; i += blockDim.x)
    s[i] = outM[OUT_M + 2 * ((size_t)i * DIM + i)];
  __syncthreads();
  for (int k = 2; k <= DIM; k <<= 1) {
    for (int j = k >> 1; j > 0; j >>= 1) {
      for (int i = threadIdx.x; i < DIM; i += blockDim.x) {
        int ixj = i ^ j;
        if (ixj > i) {
          bool up = ((i & k) == 0);
          double a = s[i], b = s[ixj];
          if ((a > b) == up) { s[i] = b; s[ixj] = a; }
        }
      }
      __syncthreads();
    }
  }
  for (int i = threadIdx.x; i < DIM; i += blockDim.x)
    out[OUT_EIG + i] = s[i];
}

// ================= WMMA complex power-iteration step: Y = M @ V =================
// V lives in ws in fragment order: frag(k,n) = ((k>>3)*16 + n)*8 + (k&7),
// Vr-frags at [0,32768), Vi-frags at [32768,65536). One TDM tensor_load_to_lds
// pulls the whole 128 KB into LDS (static LDS offset 0); B-fragments are then
// contiguous 16-byte ds_load_b128 granules.
static __device__ __forceinline__ v16h cat16(v8h lo, v8h hi) {
  v16h r;
#pragma unroll
  for (int e = 0; e < 8; ++e) { r[e] = lo[e]; r[8 + e] = hi[e]; }
  return r;
}

__global__ void k_wmma_power(const _Float16* __restrict__ Mr, const _Float16* __restrict__ Mi,
                             const _Float16* __restrict__ Vf,
                             float* __restrict__ Yr, float* __restrict__ Yi) {
  __shared__ _Float16 sV[65536];          // 128 KB: Vr-frag | Vi-frag (static LDS offset 0)

#if HAVE_TDM
  if (threadIdx.x < 32) {                 // one wave issues the DMA (EXEC ignored by TDM)
    uint64_t ga = (uint64_t)(uintptr_t)Vf;
    tdm_v4u g0;
    g0[0] = 1u;                                   // count=1, user mode, no gather
    g0[1] = 0u;                                   // lds_addr = 0 (sV is the only LDS alloc)
    g0[2] = (unsigned int)(ga & 0xFFFFFFFFu);     // global_addr[31:0]
    g0[3] = (unsigned int)((ga >> 32) & 0x1FFFFFFu) | (2u << 30);  // ga[56:32] | type=2
    tdm_v8i g1;
    g1[0] = (int)(1u << 16);                      // workgroup_mask=0, data_size=1 (2 bytes)
    g1[1] = (int)0x80000000u;                     // tensor_dim0 = 32768 (low16 -> bits[63:48])
    g1[2] = (int)(2u << 16);                      // tensor_dim0 hi=0 | tensor_dim1 = 2
    g1[3] = (int)0x80000000u;                     // tensor_dim1 hi=0 | tile_dim0 = 32768
    g1[4] = 2;                                    // tile_dim1 = 2, tile_dim2 = 0
    g1[5] = 32768;                                // tensor_dim0_stride = 32768
    g1[6] = 0;
    g1[7] = 0;
    tdm_v4i g2 = (tdm_v4i){0, 0, 0, 0};
    tdm_v4i g3 = (tdm_v4i){0, 0, 0, 0};
#if defined(__clang_major__) && (__clang_major__ >= 23)
    tdm_v8i g4 = (tdm_v8i){0, 0, 0, 0, 0, 0, 0, 0};
    __builtin_amdgcn_tensor_load_to_lds(g0, g1, g2, g3, g4, 0);
#else
    __builtin_amdgcn_tensor_load_to_lds(g0, g1, g2, g3, 0);
#endif
    __builtin_amdgcn_s_wait_tensorcnt(0);
  }
  __syncthreads();
#else
  for (int i = threadIdx.x; i < 65536 / 8; i += blockDim.x)
    ((v8h*)sV)[i] = ((const v8h*)Vf)[i];
  __syncthreads();
#endif

  // The TDM writes LDS through a descriptor the optimizer cannot see. Escape
  // sV's address into an asm with a memory clobber (so loads from sV cannot be
  // folded to undef), but keep the loads themselves on the original sV array so
  // address-space inference still emits ds_load (not flat_load).
  {
    const void* esc = (const void*)sV;
    asm volatile("" : : "v"(esc) : "memory");
  }

  const int wid  = blockIdx.x * (blockDim.x >> 5) + (threadIdx.x >> 5);
  const int lane = threadIdx.x & 31;
  const int m0   = wid << 4;
  if (m0 >= DIM) return;
  const int half = lane >> 4;
  const int mn   = lane & 15;
  const size_t rowoff = (size_t)(m0 + mn) * DIM;
  const v8h* sG = (const v8h*)sV;         // 16-byte B granules; Vi granules at +4096

  v8f acc_rr = {}, acc_ri = {}, acc_ir = {}, acc_ii = {};

#pragma unroll 2
  for (int kk = 0; kk < DIM; kk += 32) {
    const _Float16* pr = Mr + rowoff + kk + half * 8;
    const _Float16* pi = Mi + rowoff + kk + half * 8;
    v16h aR = cat16(*(const v8h*)pr, *(const v8h*)(pr + 16));
    v16h aI = cat16(*(const v8h*)pi, *(const v8h*)(pi + 16));
    if (kk + 128 < DIM) {
      __builtin_prefetch((const void*)(pr + 128), 0, 1);
      __builtin_prefetch((const void*)(pi + 128), 0, 1);
    }
    const int kb = kk >> 3;
    v16h bR = cat16(sG[(kb + half) * 16 + mn],        sG[(kb + 2 + half) * 16 + mn]);
    v16h bI = cat16(sG[4096 + (kb + half) * 16 + mn], sG[4096 + (kb + 2 + half) * 16 + mn]);
    acc_rr = __builtin_amdgcn_wmma_f32_16x16x32_f16(false, aR, false, bR, (short)0, acc_rr, false, false);
    acc_ii = __builtin_amdgcn_wmma_f32_16x16x32_f16(false, aI, false, bI, (short)0, acc_ii, false, false);
    acc_ri = __builtin_amdgcn_wmma_f32_16x16x32_f16(false, aR, false, bI, (short)0, acc_ri, false, false);
    acc_ir = __builtin_amdgcn_wmma_f32_16x16x32_f16(false, aI, false, bR, (short)0, acc_ir, false, false);
  }

#pragma unroll
  for (int r = 0; r < 8; ++r) {           // C/D layout: VGPR r -> row m0 + r + 8*half
    int m = m0 + r + 8 * half;
    Yr[m * 16 + mn] = acc_rr[r] - acc_ii[r];
    Yi[m * 16 + mn] = acc_ri[r] + acc_ir[r];
  }
}

// column norms -> rho, normalize into fragment-order f16 V
__global__ void k_power_norm(const float* __restrict__ Yr, const float* __restrict__ Yi,
                             _Float16* __restrict__ Vf, double* __restrict__ rho) {
  __shared__ double part[256];
  __shared__ double invn[16];
  __shared__ double norms[16];
  int col = threadIdx.x & 15;
  int sub = threadIdx.x >> 4;
  double acc = 0.0;
  for (int r = sub; r < DIM; r += 16) {
    double yr = Yr[r * 16 + col], yi = Yi[r * 16 + col];
    acc += yr * yr + yi * yi;
  }
  part[threadIdx.x] = acc;
  __syncthreads();
  if (sub == 0) {
    double s = 0.0;
    for (int k = 0; k < 16; ++k) s += part[k * 16 + col];
    double n = sqrt(s);
    norms[col] = n;
    invn[col] = n > 0.0 ? 1.0 / n : 0.0;
  }
  __syncthreads();
  if (threadIdx.x == 0) {
    double mx = 0.0;
    for (int k = 0; k < 16; ++k) mx = fmax(mx, norms[k]);
    *rho = mx;
  }
  for (int idx = threadIdx.x; idx < DIM * 16; idx += blockDim.x) {
    int k = idx >> 4, n = idx & 15;
    int frag = ((k >> 3) * 16 + n) * 8 + (k & 7);
    Vf[frag]         = (_Float16)(float)(Yr[idx] * invn[n]);
    Vf[32768 + frag] = (_Float16)(float)(Yi[idx] * invn[n]);
  }
}

__global__ void k_metrics(const double* __restrict__ scal, double* __restrict__ out) {
  double fro = scal[0], mn = scal[1], rho = scal[2];
  out[OUT_MET + 0] = fro;
  out[OUT_MET + 1] = rho;
  out[OUT_MET + 2] = mn > 0.0 ? rho / mn : INFINITY;
  out[OUT_MET + 3] = rho > 0.0 ? 1.0 / rho : INFINITY;
  out[OUT_MET + 4] = rho;
}

// ================= launch =================
extern "C" void kernel_launch(void* const* d_in, const int* in_sizes, int n_in,
                              void* d_out, int out_size, void* d_ws, size_t ws_size,
                              hipStream_t stream) {
  const double* arnold  = (const double*)d_in[0];
  const double* algebra = (const double*)d_in[1];
  const double* s_real  = (const double*)d_in[2];
  const double* s_imag  = (const double*)d_in[3];
  const double* theta   = (const double*)d_in[4];
  double* out = (double*)d_out;
  char*   ws  = (char*)d_ws;

  _Float16* Mr16 = (_Float16*)(ws + WS_MR16);
  _Float16* Mi16 = (_Float16*)(ws + WS_MI16);
  _Float16* Vf   = (_Float16*)(ws + WS_VFRAG);
  float*    Yr   = (float*)(ws + WS_YR);
  float*    Yi   = (float*)(ws + WS_YI);
  double* rowsum = (double*)(ws + WS_ROWSUM);
  double* scal   = (double*)(ws + WS_SCAL);

  k_zero_out<<<2048, 256, 0, stream>>>(out);
  k_ws_init<<<2048, 256, 0, stream>>>(Mr16, Mi16, Vf);
  k_build_band<<<(DIM * 15 + 255) / 256, 256, 0, stream>>>(arnold, algebra, s_real, s_imag, theta,
                                                           out, Mr16, Mi16);
  k_conn<<<(DIM * 17 + 255) / 256, 256, 0, stream>>>(out, out);
  k_row_stats<<<DIM / 256, 256, 0, stream>>>(out, rowsum);
  k_finalize_diag<<<1, 256, 0, stream>>>(out, rowsum, out, scal);
  k_sort_eig<<<1, 256, 0, stream>>>(out, out);

  for (int it = 0; it < 8; ++it) {
    k_wmma_power<<<DIM / 16 / 8, 256, 0, stream>>>(Mr16, Mi16, Vf, Yr, Yi);
    k_power_norm<<<1, 256, 0, stream>>>(Yr, Yi, Vf, scal + 2);
  }
  k_metrics<<<1, 1, 0, stream>>>(scal, out);
}
